// GlycanAtomTopologicalEncoder_22101901705608
// MI455X (gfx1250) — compile-verified, matrix operand order checked by reading, not run on hardware
//
#include <hip/hip_runtime.h>
#include <hip/hip_bf16.h>
#include <stdint.h>

// GlycanAtomTopologicalEncoder for MI455X (gfx1250, wave32).
// Strategy:
//  1) idx = argmax(one-hot atom_to_token) computed as a WMMA dot product with
//     the index vector (v_wmma_f32_16x16x32_f16, 8 accumulating WMMAs / tile).
//  2) Token-level Floyd-Warshall (256 nodes) fully LDS-resident (128KB u16
//     dist fits in the 320KB WGP LDS), packed 2xu16 min-plus per u32.
//  3) Broadcast expansion to the (4,1024,1024) int32 output with b128 stores
//     (16MB write dominates: ~0.7us at 23.3 TB/s).

typedef __attribute__((ext_vector_type(16))) _Float16 v16h;
typedef __attribute__((ext_vector_type(8)))  float    v8f;

#define BATCH   4
#define N_ATOM  1024
#define N_TOK   256
#define INF16   0x7FFFu   // INF+INF = 0xFFFE: no u16 overflow, no cross-lane carry

// ---------------------------------------------------------------------------
// Kernel 1: atom -> token index via WMMA (D = onehot x indexvec), plus glycan
// flag, packed as code = idx | (glycan ? 0x100 : 0). 256 waves, 1 tile each.
// A-fragment layout per ISA 7.12.2 (16-bit A 16x32): lane L holds M=L%16,
// half=L/16; elements e<8 -> K = kb+e, e>=8 -> K = kb+16+e-8, kb = 32*chunk+8*half.
// B K-layout mirrors A's; B depends only on K so every lane holds the K values.
// ---------------------------------------------------------------------------
__global__ void __launch_bounds__(256) glycan_idx_wmma(
    const float* __restrict__ a2t, const int* __restrict__ mono,
    unsigned short* __restrict__ code)
{
    const int tid  = threadIdx.x;
    const int lane = tid & 31;
    const int wave = (blockIdx.x * 256 + tid) >> 5;   // 0..255 global wave id
    const int b    = wave >> 6;                       // batch
    const int row0 = (wave & 63) << 4;                // first atom of 16-row tile
    const int half = lane >> 4;
    const int m    = lane & 15;

    const float* rowp = a2t + (size_t)(b * N_ATOM + row0 + m) * N_TOK;
    v8f acc = {};

#pragma unroll
    for (int chunk = 0; chunk < 8; ++chunk) {
        const int kb = chunk * 32 + half * 8;
        // two contiguous 8-float runs per lane (32B-aligned b128 loads)
        float4 f0 = *(const float4*)(rowp + kb);
        float4 f1 = *(const float4*)(rowp + kb + 4);
        float4 f2 = *(const float4*)(rowp + kb + 16);
        float4 f3 = *(const float4*)(rowp + kb + 20);
        v16h a, bv;
        a[0]  = (_Float16)f0.x; a[1]  = (_Float16)f0.y;
        a[2]  = (_Float16)f0.z; a[3]  = (_Float16)f0.w;
        a[4]  = (_Float16)f1.x; a[5]  = (_Float16)f1.y;
        a[6]  = (_Float16)f1.z; a[7]  = (_Float16)f1.w;
        a[8]  = (_Float16)f2.x; a[9]  = (_Float16)f2.y;
        a[10] = (_Float16)f2.z; a[11] = (_Float16)f2.w;
        a[12] = (_Float16)f3.x; a[13] = (_Float16)f3.y;
        a[14] = (_Float16)f3.z; a[15] = (_Float16)f3.w;
#pragma unroll
        for (int e = 0; e < 8; ++e) {
            bv[e]     = (_Float16)(short)(kb + e);        // K value itself
            bv[e + 8] = (_Float16)(short)(kb + 16 + e);
        }
        // D = A x B + C  (emits v_wmma_f32_16x16x32_f16)
        acc = __builtin_amdgcn_wmma_f32_16x16x32_f16(
            false, a, false, bv, (short)0, acc, false, false);
    }

    // C/D layout: lane L column N=L%16, VGPR r holds M = r + 8*(L/16).
    // Every column holds the same value = idx; column-0 lanes write out.
    if (m == 0) {
#pragma unroll
        for (int r = 0; r < 8; ++r) {
            const int atom = row0 + half * 8 + r;
            const int t    = (int)(acc[r] + 0.5f);        // exact integer sum
            const int g    = (mono[b * N_ATOM + atom] != -1) ? 0x100 : 0;
            code[b * N_ATOM + atom] = (unsigned short)(t | g);
        }
    }
}

// ---------------------------------------------------------------------------
// Kernel 2: token-level Floyd-Warshall in LDS. One 1024-thread block / batch.
// Dynamic LDS: u16 dist[256][256] (128KB) + u32 usable[256] (1KB).
// Thread owns (row i = tid>>2, 64-col strip). Row/col k are invariant inside
// step k, so a single barrier per k suffices. Non-usable k are skipped.
// ---------------------------------------------------------------------------
__global__ void __launch_bounds__(1024) glycan_token_fw(
    const float* __restrict__ token_bonds,
    const unsigned short* __restrict__ code,
    unsigned short* __restrict__ distT,
    unsigned short* __restrict__ selfd)
{
    extern __shared__ char smem[];
    unsigned short* dist  = (unsigned short*)smem;                      // 256*256 u16
    unsigned int* usable  = (unsigned int*)(smem + N_TOK * N_TOK * 2);  // 256 u32
    const int tid = threadIdx.x;
    const int b   = blockIdx.x;

    if (tid < N_TOK) usable[tid] = 0u;
    __syncthreads();
    {   // scatter: token usable iff it contains >=1 glycan atom
        const unsigned int c = code[b * N_ATOM + tid];
        if (c & 0x100u) atomicOr(&usable[c & 0xFFu], 1u);
    }
    __syncthreads();

    const int i  = tid >> 2;
    const int j0 = (tid & 3) * 64;
    const float* tbrow = token_bonds + (size_t)(b * N_TOK + i) * N_TOK;
    const unsigned int ui = usable[i];
    for (int j = j0; j < j0 + 64; ++j) {
        unsigned short v;
        if (i == j)
            v = ui ? (unsigned short)0 : (unsigned short)INF16;
        else
            v = (ui && usable[j] && (tbrow[j] > 0.0f)) ? (unsigned short)1
                                                       : (unsigned short)INF16;
        dist[i * N_TOK + j] = v;
    }
    __syncthreads();

    // selfdist[t]: distance between two distinct glycan atoms of token t:
    // 1 if self-bond, else 2 if any usable neighbor (out-and-back), else INF.
    if (tid < N_TOK) {
        unsigned short sd = (unsigned short)INF16;
        if (usable[tid]) {
            if (token_bonds[(size_t)(b * N_TOK + tid) * N_TOK + tid] > 0.0f) {
                sd = 1;
            } else {
                const unsigned int* r = (const unsigned int*)&dist[tid * N_TOK];
                for (int c = 0; c < N_TOK / 2; ++c) {
                    const unsigned int v = r[c];
                    if ((v & 0xFFFFu) == 1u || (v >> 16) == 1u) { sd = 2; break; }
                }
            }
        }
        selfd[b * N_TOK + tid] = sd;
    }

    // Floyd-Warshall, packed 2xu16 min-plus per u32 (no cross-lane carry:
    // both addends <= 0x7FFF so each 16-bit lane sum <= 0xFFFE).
    unsigned int* drow = (unsigned int*)&dist[i * N_TOK + j0];
    for (int k = 0; k < N_TOK; ++k) {
        __syncthreads();
        if (!usable[k]) continue;                   // uniform: row/col k all INF
        const unsigned int dik = dist[i * N_TOK + k];
        if (dik >= INF16) continue;                 // nothing to relax via k
        const unsigned int dik2 = dik * 0x00010001u;
        const unsigned int* rowk = (const unsigned int*)&dist[k * N_TOK + j0];
#pragma unroll 8
        for (int c = 0; c < 32; ++c) {
            const unsigned int cur = drow[c];
            const unsigned int sum = rowk[c] + dik2;
            const unsigned int lo  = min(cur & 0xFFFFu, sum & 0xFFFFu);
            const unsigned int hi  = min(cur >> 16,     sum >> 16);
            drow[c] = lo | (hi << 16);
        }
    }
    __syncthreads();

    // spill token distances (L2-resident, 512KB total across batches)
    const unsigned int* src = (const unsigned int*)dist;
    unsigned int* dst = (unsigned int*)(distT + (size_t)b * N_TOK * N_TOK);
    for (int c = tid; c < N_TOK * N_TOK / 2; c += 1024) dst[c] = src[c];
}

// ---------------------------------------------------------------------------
// Kernel 3: expansion to (B,1024,1024) int32. Pure write bandwidth: 4 outputs
// per thread, b128 stores. distT rows stay hot in cache.
// ---------------------------------------------------------------------------
__global__ void __launch_bounds__(256) glycan_expand(
    const unsigned short* __restrict__ code,
    const unsigned short* __restrict__ distT,
    const unsigned short* __restrict__ selfd,
    int* __restrict__ out)
{
    const int flat = blockIdx.x * 256 + threadIdx.x;   // < 4*1024*256
    const int b  = flat >> 18;
    const int r  = flat & 0x3FFFF;
    const int i  = r >> 8;
    const int j0 = (r & 255) << 2;

    const unsigned int ci = code[(b << 10) + i];
    const int  ti = (int)(ci & 0xFFu);
    const bool gi = (ci & 0x100u) != 0u;
    const unsigned short* drow = distT + (((size_t)(b << 8) + ti) << 8);
    __builtin_prefetch(drow, 0, 1);                    // global_prefetch
    const unsigned short sdv = selfd[(b << 8) + ti];
    const unsigned short* cj = code + (b << 10) + j0;  // 8B-aligned run of 4

    int res[4];
#pragma unroll
    for (int jj = 0; jj < 4; ++jj) {
        const int j = j0 + jj;
        const unsigned int c = cj[jj];
        int v;
        if (i == j) {
            v = 0;                                     // diagonal forced to 0
        } else if (!gi || !(c & 0x100u)) {
            v = -1;                                    // non-glycan pair
        } else {
            const int tj = (int)(c & 0xFFu);
            const unsigned short d = (tj == ti) ? sdv : drow[tj];
            v = (d < INF16) ? (int)d : -1;
        }
        res[jj] = v;
    }
    *(int4*)(out + (((size_t)b << 20) + ((size_t)i << 10) + (size_t)j0)) =
        make_int4(res[0], res[1], res[2], res[3]);
}

// ---------------------------------------------------------------------------
extern "C" void kernel_launch(void* const* d_in, const int* in_sizes, int n_in,
                              void* d_out, int out_size, void* d_ws, size_t ws_size,
                              hipStream_t stream) {
    (void)in_sizes; (void)n_in; (void)out_size; (void)ws_size;
    // d_in[0] = atom_pad_mask (unused by the reference beyond its shape)
    const int*   mono = (const int*)  d_in[1];   // atom_mono_idx  (B,1024) i32
    const float* tb   = (const float*)d_in[2];   // token_bonds    (B,256,256,1) f32
    const float* a2t  = (const float*)d_in[3];   // atom_to_token  (B,1024,256) f32
    int* out = (int*)d_out;                      // reference output dtype: int32

    // workspace layout (~534KB)
    unsigned short* code  = (unsigned short*)d_ws;                          // 8KB
    unsigned short* distT = (unsigned short*)((char*)d_ws + 8192);          // 512KB
    unsigned short* selfd = (unsigned short*)((char*)d_ws + 8192 +
                                              (size_t)BATCH * N_TOK * N_TOK * 2); // 2KB

    glycan_idx_wmma<<<32, 256, 0, stream>>>(a2t, mono, code);

    const size_t lds = (size_t)N_TOK * N_TOK * 2 + (size_t)N_TOK * 4; // 132KB < 320KB WGP LDS
    glycan_token_fw<<<BATCH, 1024, lds, stream>>>(tb, code, distT, selfd);

    glycan_expand<<<(BATCH * N_ATOM * N_TOK) / 256, 256, 0, stream>>>(
        code, distT, selfd, out);
}
